// MultiHeadAttention_43825846289057
// MI455X (gfx1250) — compile-verified, matrix operand order checked by reading, not run on hardware
//
#include <hip/hip_runtime.h>
#include <hip/hip_bf16.h>

// MHA forward for B=4, L=2048, D=1024, H=16, DK=DV=64 on gfx1250.
// All GEMMs use v_wmma_f32_16x16x32_f16 (wave32 WMMA).
// Fused scores+softmax+PV avoids re-reading the 1.07GB attn matrix from HBM.
// Projection kernels stage shared weight fragments block-wide via the
// Tensor Data Mover (tensor_load_to_lds + s_wait_tensorcnt), double-buffered.

typedef __attribute__((ext_vector_type(16))) _Float16 v16h;
typedef __attribute__((ext_vector_type(8)))  float    v8f;
typedef __attribute__((ext_vector_type(4)))  float    v4f;
typedef __attribute__((ext_vector_type(4)))  unsigned int v4u;
typedef __attribute__((ext_vector_type(8)))  int      v8i;
typedef __attribute__((ext_vector_type(4)))  int      v4i;

#define WMMA_F16(a, b, c) \
  __builtin_amdgcn_wmma_f32_16x16x32_f16(false, (a), false, (b), (short)0, (c), false, false)

#define HAVE_TDM __has_builtin(__builtin_amdgcn_tensor_load_to_lds)

static constexpr int Bb = 4, Ll = 2048, Dd = 1024, Hh = 16, DKk = 64;
static constexpr float MINV = -3.402823466e+38f;

// Load 16 consecutive fp32, convert to a v16h fragment half-row.
__device__ inline v16h load_cvt_a(const float* __restrict__ p) {
  v16h a;
#pragma unroll
  for (int i = 0; i < 16; i += 4) {
    v4f f = *(const v4f*)(p + i);
    a[i + 0] = (_Float16)f[0];
    a[i + 1] = (_Float16)f[1];
    a[i + 2] = (_Float16)f[2];
    a[i + 3] = (_Float16)f[3];
  }
  return a;
}

#if HAVE_TDM
// Issue a TDM 2-D tile load: 4 rows x 4096B (row stride 32768B) -> LDS.
// D# per CDNA5 ISA ch.8: group0 {count, lds_addr, global_addr, type=2},
// group1 {data_size=8B, tensor 512x4 (8B units), tile 512x4, stride 4096}.
__device__ inline void tdm_load_wtile(const _Float16* g, void* lds) {
  unsigned long long ga = (unsigned long long)(size_t)g;
  unsigned la = (unsigned)(size_t)lds;   // low 32 bits of flat LDS addr
  v4u g0;
  g0[0] = 1u;                                  // count = 1 (valid descriptor)
  g0[1] = la;                                  // lds_addr
  g0[2] = (unsigned)ga;                        // global_addr[31:0]
  g0[3] = (unsigned)(ga >> 32) | 0x80000000u;  // global_addr[56:32] | type=2
  v8i g1;
  g1[0] = (3 << 16);     // workgroup_mask=0, data_size=3 (8 bytes)
  g1[1] = (512 << 16);   // abar_addr=0, tensor_dim0[15:0]=512
  g1[2] = (4 << 16);     // tensor_dim0 hi=0, tensor_dim1[15:0]=4
  g1[3] = (512 << 16);   // tensor_dim1 hi=0, tile_dim0=512
  g1[4] = 4;             // tile_dim1=4, tile_dim2=0
  g1[5] = 4096;          // tensor_dim0_stride[31:0]=4096 (x8B = 32KB)
  g1[6] = 0;             // stride hi, tensor_dim1_stride lo
  g1[7] = 0;
  v4i z4 = {0, 0, 0, 0};
  v8i z8 = {0, 0, 0, 0, 0, 0, 0, 0};
  __builtin_amdgcn_tensor_load_to_lds(g0, g1, z4, z4, z8, 0);
}
#endif

// ---------------------------------------------------------------------------
// Repack fp32 weight [K,N] (row-major) into f16 B-fragment layout:
// P[((n_tile*(K/32) + k_blk)*32 + lane)*16 + e] = W[k, n]
// ---------------------------------------------------------------------------
__global__ void repack_w(const float* __restrict__ W, _Float16* __restrict__ P,
                         int K, int N) {
  int p = blockIdx.x * blockDim.x + threadIdx.x;
  if (p >= K * N) return;
  int frag   = p >> 9;          // 512 halves per fragment
  int within = p & 511;
  int lane   = within >> 4;
  int e      = within & 15;
  int kblocks = K >> 5;
  int n_tile  = frag / kblocks;
  int k_blk   = frag % kblocks;
  int k = k_blk * 32 + ((lane >> 4) << 4) + e;
  int n = n_tile * 16 + (lane & 15);
  P[p] = (_Float16)W[k * N + n];
}

// ---------------------------------------------------------------------------
// Fused QKV projection. grid = (64, 16, 3), block = 256 (8 waves).
// Weight fragments (identical across the 8 waves) are staged block-wide via
// TDM into double-buffered LDS (2 x 16KB = 4 n-tiles x 4 k-steps each);
// s_wait_tensorcnt 1 overlaps the next DMA with the current 16 WMMAs/wave.
// Outputs: qh/kh f16 [B,H,L,64]; vt f16 [B,H,64,L] (transposed for P@V).
// ---------------------------------------------------------------------------
__global__ void qkv_proj(const float* __restrict__ q, const float* __restrict__ k,
                         const float* __restrict__ v,
                         const _Float16* __restrict__ wq, const _Float16* __restrict__ wk,
                         const _Float16* __restrict__ wv,
                         const float* __restrict__ bq, const float* __restrict__ bk,
                         const float* __restrict__ bv,
                         _Float16* __restrict__ qh, _Float16* __restrict__ kh,
                         _Float16* __restrict__ vt) {
  const int lane = threadIdx.x & 31;
  const int wave = threadIdx.x >> 5;
  const int lo = lane & 15, hi = lane >> 4;
  const int m0 = blockIdx.x * 128 + wave * 16;
  const int n0 = blockIdx.y * 64;
  const int which = blockIdx.z;

  const float* x     = (which == 0) ? q  : (which == 1) ? k  : v;
  const _Float16* wp = (which == 0) ? wq : (which == 1) ? wk : wv;
  const float* bias  = (which == 0) ? bq : (which == 1) ? bk : bv;

  v8f acc[4] = {};
  const float* arow = x + (m0 + lo) * Dd + (hi << 4);

#if HAVE_TDM
  __shared__ _Float16 wtile[2][4 * 4 * 512];   // 2 x 16KB
  // fragments for (n_tile0+j, kb): wp + ((n_tile0+j)*32 + kb)*512
  const _Float16* wbase = wp + (size_t)(n0 >> 4) * 32 * 512;
  if (wave == 0) tdm_load_wtile(wbase, &wtile[0][0]);   // macro-step 0
  for (int ks = 0; ks < 8; ++ks) {
    if (wave == 0) {
      if (ks + 1 < 8) {
        tdm_load_wtile(wbase + (ks + 1) * 4 * 512, &wtile[(ks + 1) & 1][0]);
        __builtin_amdgcn_s_wait_tensorcnt(1);   // oldest (ks) complete
      } else {
        __builtin_amdgcn_s_wait_tensorcnt(0);   // final tile complete
      }
    }
    __syncthreads();                    // buf[ks&1] ready for all waves
    const _Float16* wb = &wtile[ks & 1][0];
#pragma unroll
    for (int ki = 0; ki < 4; ++ki) {
      int kb = ks * 4 + ki;
      v16h a = load_cvt_a(arow + kb * 32);
#pragma unroll
      for (int j = 0; j < 4; ++j) {
        v16h bf = *(const v16h*)(wb + (j * 4 + ki) * 512 + lane * 16);
        acc[j] = WMMA_F16(a, bf, acc[j]);
      }
    }
    __syncthreads();                    // consumption done; safe to overwrite
  }
#else
#pragma unroll 4
  for (int kb = 0; kb < 32; ++kb) {
    v16h a = load_cvt_a(arow + kb * 32);
#pragma unroll
    for (int j = 0; j < 4; ++j) {
      const _Float16* bp = wp + ((((n0 >> 4) + j) * 32 + kb) * 32 + lane) * 16;
      v16h bf = *(const v16h*)bp;
      acc[j] = WMMA_F16(a, bf, acc[j]);
    }
  }
#endif

  const int mbase = m0 + (hi << 3);
#pragma unroll
  for (int j = 0; j < 4; ++j) {
    int n = n0 + j * 16 + lo;
    float bia = bias[n];
    int h = n >> 6, d = n & 63;
#pragma unroll
    for (int r = 0; r < 8; ++r) {
      int m = mbase + r;              // global row = b*L + l
      int b = m >> 11, l = m & 2047;
      float val = acc[j][r] + bia;
      if (which == 2) {
        vt[((b * Hh + h) * DKk + d) * Ll + l] = (_Float16)val;
      } else {
        _Float16* o = (which == 0) ? qh : kh;
        o[((b * Hh + h) * Ll + l) * DKk + d] = (_Float16)val;
      }
    }
  }
}

// ---------------------------------------------------------------------------
// Fused attention: scores -> mask -> softmax -> write attn (NT) -> P@V.
// grid = (L/16, B*H), block = 256 (8 waves).
// ---------------------------------------------------------------------------
__global__ void attn_fused(const _Float16* __restrict__ qh,
                           const _Float16* __restrict__ kh,
                           const int* __restrict__ mask,
                           const _Float16* __restrict__ vt,
                           float* __restrict__ attn,
                           _Float16* __restrict__ oh) {
  __shared__ _Float16 probs[16 * Ll];     // 64KB: 16 query rows x 2048 keys
  __shared__ float acc_sh[16 * DKk];      // 4KB : 16 x 64 output tile
  __shared__ float red[8][16];

  const int lane = threadIdx.x & 31;
  const int wave = threadIdx.x >> 5;
  const int lo = lane & 15, hi = lane >> 4;
  const int q0 = blockIdx.x * 16;
  const int bh = blockIdx.y;          // b*H + h
  const int b = bh >> 4, h = bh & 15;
  const int kbase = wave * 256;

  // zero the output accumulation tile (visible after the staging barrier)
#pragma unroll
  for (int i = 0; i < 4; ++i) acc_sh[threadIdx.x * 4 + i] = 0.f;

  // prefetch this wave's mask slice (16 rows x 256 keys) to overlap WMMAs
#pragma unroll
  for (int r = 0; r < 8; ++r)
    __builtin_prefetch(&mask[(b * Ll + q0 + r + (hi << 3)) * Ll + kbase + lo * 16], 0, 1);

  // ---- Phase 1: scores ----
  v8f s[16] = {};
  const _Float16* qrow = qh + (bh * Ll + q0 + lo) * DKk + (hi << 4);
#pragma unroll
  for (int ds = 0; ds < 2; ++ds) {
    v16h a = *(const v16h*)(qrow + ds * 32);
#pragma unroll
    for (int t = 0; t < 16; ++t) {
      const _Float16* kp = kh + (bh * Ll + kbase + t * 16 + lo) * DKk + ds * 32 + (hi << 4);
      v16h bf = *(const v16h*)kp;
      s[t] = WMMA_F16(a, bf, s[t]);
    }
  }

  // scale by 1/sqrt(64) and apply mask
#pragma unroll
  for (int t = 0; t < 16; ++t) {
#pragma unroll
    for (int r = 0; r < 8; ++r) {
      int m  = q0 + r + (hi << 3);
      int kc = kbase + t * 16 + lo;
      int mv = mask[(b * Ll + m) * Ll + kc];
      float val = s[t][r] * 0.125f;
      s[t][r] = (mv == 0) ? MINV : val;
    }
  }

  // ---- softmax: row max ----
  float gmax[8];
  {
    float rmax[8];
#pragma unroll
    for (int r = 0; r < 8; ++r) {
      float mx = s[0][r];
#pragma unroll
      for (int t = 1; t < 16; ++t) mx = fmaxf(mx, s[t][r]);
#pragma unroll
      for (int off = 8; off >= 1; off >>= 1)
        mx = fmaxf(mx, __shfl_xor(mx, off, 32));
      rmax[r] = mx;
    }
    if (lo == 0) {
#pragma unroll
      for (int r = 0; r < 8; ++r) red[wave][(hi << 3) + r] = rmax[r];
    }
    __syncthreads();
#pragma unroll
    for (int r = 0; r < 8; ++r) {
      float mx = red[0][(hi << 3) + r];
#pragma unroll
      for (int w = 1; w < 8; ++w) mx = fmaxf(mx, red[w][(hi << 3) + r]);
      gmax[r] = mx;
    }
    __syncthreads();
  }

  // ---- exp + row sum ----
  float rsum[8];
#pragma unroll
  for (int r = 0; r < 8; ++r) rsum[r] = 0.f;
#pragma unroll
  for (int t = 0; t < 16; ++t) {
#pragma unroll
    for (int r = 0; r < 8; ++r) {
      float e = __expf(s[t][r] - gmax[r]);
      s[t][r] = e;
      rsum[r] += e;
    }
  }
#pragma unroll
  for (int r = 0; r < 8; ++r) {
    float sm = rsum[r];
#pragma unroll
    for (int off = 8; off >= 1; off >>= 1) sm += __shfl_xor(sm, off, 32);
    rsum[r] = sm;
  }
  if (lo == 0) {
#pragma unroll
    for (int r = 0; r < 8; ++r) red[wave][(hi << 3) + r] = rsum[r];
  }
  __syncthreads();
  float ginv[8];
#pragma unroll
  for (int r = 0; r < 8; ++r) {
    float sm = 0.f;
#pragma unroll
    for (int w = 0; w < 8; ++w) sm += red[w][(hi << 3) + r];
    ginv[r] = 1.0f / sm;
  }

  // ---- write attn (fp32, nontemporal) + stage probs (f16) in LDS ----
#pragma unroll
  for (int t = 0; t < 16; ++t) {
#pragma unroll
    for (int r = 0; r < 8; ++r) {
      int mloc = r + (hi << 3);
      int kc   = kbase + t * 16 + lo;
      float p  = s[t][r] * ginv[r];
      __builtin_nontemporal_store(p, &attn[(bh * Ll + q0 + mloc) * Ll + kc]);
      probs[mloc * Ll + kc] = (_Float16)p;
    }
  }
  __syncthreads();

  // ---- Phase 2: P @ V, k split across waves ----
  v8f acc[4] = {};
  for (int kb = 0; kb < 8; ++kb) {
    int k0 = kbase + kb * 32;
    v16h a = *(const v16h*)(&probs[lo * Ll + k0 + (hi << 4)]);
#pragma unroll
    for (int j = 0; j < 4; ++j) {
      const _Float16* vp = vt + (bh * DKk + j * 16 + lo) * Ll + k0 + (hi << 4);
      v16h bf = *(const v16h*)vp;
      acc[j] = WMMA_F16(a, bf, acc[j]);
    }
  }
#pragma unroll
  for (int j = 0; j < 4; ++j) {
#pragma unroll
    for (int r = 0; r < 8; ++r) {
      int mloc = r + (hi << 3);
      atomicAdd(&acc_sh[mloc * DKk + j * 16 + lo], acc[j][r]);
    }
  }
  __syncthreads();

  // ---- store out_h (f16) [B, L, H*DV]; 1024 values, 4 per thread ----
#pragma unroll
  for (int i = 0; i < 4; ++i) {
    int idx = threadIdx.x * 4 + i;
    int mloc = idx >> 6, d = idx & 63;
    oh[(b * Ll + q0 + mloc) * Dd + h * DKk + d] = (_Float16)acc_sh[idx];
  }
}

// ---------------------------------------------------------------------------
// Final projection: out = out_h @ Wo + bo (fp32 out). grid = (64,16), blk 256.
// Same TDM-staged weight pipeline as qkv_proj.
// ---------------------------------------------------------------------------
__global__ void out_proj(const _Float16* __restrict__ oh,
                         const _Float16* __restrict__ wo,
                         const float* __restrict__ bo,
                         float* __restrict__ out) {
  const int lane = threadIdx.x & 31;
  const int wave = threadIdx.x >> 5;
  const int lo = lane & 15, hi = lane >> 4;
  const int m0 = blockIdx.x * 128 + wave * 16;
  const int n0 = blockIdx.y * 64;

  v8f acc[4] = {};
  const _Float16* arow = oh + (m0 + lo) * Dd + (hi << 4);

#if HAVE_TDM
  __shared__ _Float16 wtile[2][4 * 4 * 512];   // 2 x 16KB
  const _Float16* wbase = wo + (size_t)(n0 >> 4) * 32 * 512;
  if (wave == 0) tdm_load_wtile(wbase, &wtile[0][0]);
  for (int ks = 0; ks < 8; ++ks) {
    if (wave == 0) {
      if (ks + 1 < 8) {
        tdm_load_wtile(wbase + (ks + 1) * 4 * 512, &wtile[(ks + 1) & 1][0]);
        __builtin_amdgcn_s_wait_tensorcnt(1);
      } else {
        __builtin_amdgcn_s_wait_tensorcnt(0);
      }
    }
    __syncthreads();
    const _Float16* wb = &wtile[ks & 1][0];
#pragma unroll
    for (int ki = 0; ki < 4; ++ki) {
      int kb = ks * 4 + ki;
      v16h a = *(const v16h*)(arow + kb * 32);
#pragma unroll
      for (int j = 0; j < 4; ++j) {
        v16h bf = *(const v16h*)(wb + (j * 4 + ki) * 512 + lane * 16);
        acc[j] = WMMA_F16(a, bf, acc[j]);
      }
    }
    __syncthreads();
  }
#else
#pragma unroll 4
  for (int kb = 0; kb < 32; ++kb) {
    v16h a = *(const v16h*)(arow + kb * 32);
#pragma unroll
    for (int j = 0; j < 4; ++j) {
      const _Float16* bp = wo + ((((n0 >> 4) + j) * 32 + kb) * 32 + lane) * 16;
      v16h bf = *(const v16h*)bp;
      acc[j] = WMMA_F16(a, bf, acc[j]);
    }
  }
#endif

  const int mbase = m0 + (hi << 3);
#pragma unroll
  for (int j = 0; j < 4; ++j) {
    int n = n0 + j * 16 + lo;
    float bia = bo[n];
#pragma unroll
    for (int r = 0; r < 8; ++r) {
      out[(mbase + r) * Dd + n] = acc[j][r] + bia;
    }
  }
}

// ---------------------------------------------------------------------------
extern "C" void kernel_launch(void* const* d_in, const int* in_sizes, int n_in,
                              void* d_out, int out_size, void* d_ws, size_t ws_size,
                              hipStream_t stream) {
  (void)in_sizes; (void)n_in; (void)out_size; (void)ws_size;
  const float* q    = (const float*)d_in[0];
  const float* k    = (const float*)d_in[1];
  const float* v    = (const float*)d_in[2];
  const int*   mask = (const int*)d_in[3];
  const float* Wq   = (const float*)d_in[4];
  const float* bq   = (const float*)d_in[5];
  const float* Wk   = (const float*)d_in[6];
  const float* bk   = (const float*)d_in[7];
  const float* Wv   = (const float*)d_in[8];
  const float* bv   = (const float*)d_in[9];
  const float* Wo   = (const float*)d_in[10];
  const float* bo   = (const float*)d_in[11];

  float* out_ptr  = (float*)d_out;                        // [B, L, D]
  float* attn_ptr = (float*)d_out + (size_t)Bb * Ll * Dd; // [B, H, L, L]

  // Workspace layout (f16 elements): 4x2MB packed weights + 4x16MB tensors
  const size_t WPACK = (size_t)Dd * Dd;
  const size_t QHSZ  = (size_t)Bb * Hh * Ll * DKk;
  _Float16* wqp = (_Float16*)d_ws;
  _Float16* wkp = wqp + WPACK;
  _Float16* wvp = wkp + WPACK;
  _Float16* wop = wvp + WPACK;
  _Float16* qh  = wop + WPACK;
  _Float16* kh  = qh + QHSZ;
  _Float16* vt  = kh + QHSZ;
  _Float16* oh  = vt + QHSZ;   // total 72 MB

  const int rp_blocks = (int)((WPACK + 255) / 256);
  repack_w<<<rp_blocks, 256, 0, stream>>>(Wq, wqp, Dd, Dd);
  repack_w<<<rp_blocks, 256, 0, stream>>>(Wk, wkp, Dd, Dd);
  repack_w<<<rp_blocks, 256, 0, stream>>>(Wv, wvp, Dd, Dd);
  repack_w<<<rp_blocks, 256, 0, stream>>>(Wo, wop, Dd, Dd);

  qkv_proj<<<dim3(64, 16, 3), 256, 0, stream>>>(q, k, v, wqp, wkp, wvp,
                                                bq, bk, bv, qh, kh, vt);

  attn_fused<<<dim3(Ll / 16, Bb * Hh), 256, 0, stream>>>(qh, kh, mask, vt,
                                                         attn_ptr, oh);

  out_proj<<<dim3(64, 16), 256, 0, stream>>>(oh, wop, bo, out_ptr);
}
